// MultiHeadSelfAttention_15272903705120
// MI455X (gfx1250) — matrix operand order repaired and checked by reading, not compiled, not run on hardware
//
#include <hip/hip_runtime.h>

// ---- problem constants ----
#define BB 4
#define TT 2048
#define CC 1024
#define HH 16
#define DH 64
#define BT (BB*TT)         // 8192
#define C3 (3*CC)          // 3072

typedef __attribute__((ext_vector_type(16))) _Float16 v16h;
typedef __attribute__((ext_vector_type(8)))  _Float16 h8;
typedef __attribute__((ext_vector_type(8)))  float    v8f;

static __device__ __forceinline__ v8f wmma16(v16h a, v16h b, v8f c) {
    return __builtin_amdgcn_wmma_f32_16x16x32_f16(
        false, a, false, b, (short)0, c, false, false);
}

// A fragment: 16x32 f16 (M x K), row-major source, ld = row stride (elems).
// ISA layout: lane m (0-15) holds K = {lh*8..lh*8+7, 16+lh*8..16+lh*8+7}
static __device__ __forceinline__ v16h load_a16(const _Float16* __restrict__ t, int ld) {
    int lane = threadIdx.x & 31;
    int m = lane & 15, lh = lane >> 4;
    const _Float16* p = t + m * ld + lh * 8;
    h8 lo = *(const h8*)(p);
    h8 hi = *(const h8*)(p + 16);
    v16h r;
#pragma unroll
    for (int j = 0; j < 8; ++j) { r[j] = lo[j]; r[j + 8] = hi[j]; }
    return r;
}

// B fragment: 32x16 f16 (K x N). Source is B^T row-major (row n contiguous in K),
// ld = row stride. ISA layout: lane n (0-15) holds K = {lh*8..lh*8+7, 16+lh*8..16+lh*8+7}.
static __device__ __forceinline__ v16h load_b16(const _Float16* __restrict__ t, int ld) {
    int lane = threadIdx.x & 31;
    int n = lane & 15, lh = lane >> 4;
    const _Float16* p = t + n * ld + lh * 8;
    h8 lo = *(const h8*)(p);
    h8 hi = *(const h8*)(p + 16);
    v16h r;
#pragma unroll
    for (int j = 0; j < 8; ++j) { r[j] = lo[j]; r[j + 8] = hi[j]; }
    return r;
}

// one k-step fragment set for a 32(M) x 64(N) tile
struct FragSet { v16h a0, a1, b0, b1, b2, b3; };

static __device__ __forceinline__ FragSet
load_set(const _Float16* __restrict__ arow, const _Float16* __restrict__ brow, int k) {
    FragSet s;
    s.a0 = load_a16(arow + k, CC);
    s.a1 = load_a16(arow + 16 * CC + k, CC);
    s.b0 = load_b16(brow + 0 * 16 * CC + k, CC);
    s.b1 = load_b16(brow + 1 * 16 * CC + k, CC);
    s.b2 = load_b16(brow + 2 * 16 * CC + k, CC);
    s.b3 = load_b16(brow + 3 * 16 * CC + k, CC);
    return s;
}

#define GEMM_STEP(S)                              \
    acc[0] = wmma16((S).a0, (S).b0, acc[0]);      \
    acc[1] = wmma16((S).a0, (S).b1, acc[1]);      \
    acc[2] = wmma16((S).a0, (S).b2, acc[2]);      \
    acc[3] = wmma16((S).a0, (S).b3, acc[3]);      \
    acc[4] = wmma16((S).a1, (S).b0, acc[4]);      \
    acc[5] = wmma16((S).a1, (S).b1, acc[5]);      \
    acc[6] = wmma16((S).a1, (S).b2, acc[6]);      \
    acc[7] = wmma16((S).a1, (S).b3, acc[7]);

// ---------------- Stage 0: fp32 -> f16 convert ----------------
__global__ void cvt_f32_f16(const float* __restrict__ in, _Float16* __restrict__ out, int n) {
    int i = blockIdx.x * blockDim.x + threadIdx.x;
    if (i < n) out[i] = (_Float16)in[i];
}

// ---------------- Stage 1: QKV GEMM ----------------
// 32x64 tile per wave, ping-pong double buffered k-loop (no rotation copies).
// x[BT,C] (f16) * Wqkv[3C,C]^T -> Q[B,H,T,Dh]*scale, K[B,H,T,Dh], V^T[B,H,Dh,T]
__global__ void __launch_bounds__(256, 1)
qkv_gemm(const _Float16* __restrict__ xh,
         const _Float16* __restrict__ wh,
         _Float16* __restrict__ Qh,
         _Float16* __restrict__ Kh,
         _Float16* __restrict__ Vt) {
    int wid = blockIdx.x * 8 + (threadIdx.x >> 5);
    int mtile = wid & 255;          // BT/32 = 256
    int n64   = wid >> 8;           // 0..47 (3072/64)
    const _Float16* arow = xh + (size_t)(mtile * 32) * CC;
    const _Float16* brow = wh + (size_t)(n64 * 64) * CC;

    v8f acc[8];
#pragma unroll
    for (int i = 0; i < 8; ++i) acc[i] = (v8f){};

    FragSet s0 = load_set(arow, brow, 0);
#pragma unroll 1
    for (int k0 = 0; k0 < CC - 64; k0 += 64) {
        FragSet s1 = load_set(arow, brow, k0 + 32);
        GEMM_STEP(s0);
        s0 = load_set(arow, brow, k0 + 64);
        GEMM_STEP(s1);
    }
    {   // epilogue: k = CC-64 and CC-32
        FragSet s1 = load_set(arow, brow, CC - 32);
        GEMM_STEP(s0);
        GEMM_STEP(s1);
    }

    int lane = threadIdx.x & 31, n = lane & 15, lh = lane >> 4;
    int sec = n64 >> 4;             // 0=Q 1=K 2=V
    int h   = n64 & 15;             // 64-wide tile == exactly one head
    float scale = (sec == 0) ? 0.125f : 1.0f;   // Dh^-0.5 folded into Q
#pragma unroll
    for (int mi = 0; mi < 2; ++mi) {
#pragma unroll
        for (int nt = 0; nt < 4; ++nt) {
            int d = nt * 16 + n;
#pragma unroll
            for (int r = 0; r < 8; ++r) {
                int m = mtile * 32 + mi * 16 + r + 8 * lh;
                int b = m >> 11, t = m & (TT - 1);
                _Float16 v = (_Float16)(acc[mi * 4 + nt][r] * scale);
                size_t rowc = ((size_t)(b * HH + h) * TT + t) * DH + d;
                if (sec == 0)      Qh[rowc] = v;
                else if (sec == 1) Kh[rowc] = v;
                else Vt[((size_t)(b * HH + h) * DH + d) * TT + t] = v;
            }
        }
    }
}

// ---------------- Stage 2: causal flash attention ----------------
// per wave: 16 queries x Dh=64, online softmax over 32-key tiles
__global__ void __launch_bounds__(256, 1)
attn_fwd(const _Float16* __restrict__ Qh,
         const _Float16* __restrict__ Kh,
         const _Float16* __restrict__ Vt,
         _Float16* __restrict__ Oh) {
    __shared__ __align__(16) _Float16 ptile[8][16 * 32];
    int wv  = threadIdx.x >> 5;
    int wid = blockIdx.x * 8 + wv;
    int qt  = wid & 127;            // T/16 = 128
    int bh  = wid >> 7;             // 0..63
    int q0  = qt * 16;
    int lane = threadIdx.x & 31, n = lane & 15, lh = lane >> 4;

    const _Float16* Qb = Qh + ((size_t)bh * TT + q0) * DH;
    v16h a0 = load_a16(Qb, DH);       // Dh 0..31
    v16h a1 = load_a16(Qb + 32, DH);  // Dh 32..63

    v8f o0 = {}, o1 = {}, o2 = {}, o3 = {};
    float mrow[8], lrow[8];
#pragma unroll
    for (int r = 0; r < 8; ++r) { mrow[r] = -1e30f; lrow[r] = 0.0f; }

    _Float16* myp = &ptile[wv][0];
    int nk = (q0 + 16 + 31) >> 5;     // key tiles covering 0..q0+15
    for (int jt = 0; jt < nk; ++jt) {
        int t0 = jt * 32;
        const _Float16* Kb = Kh + ((size_t)bh * TT + t0) * DH;
        // prefetch next key tile into cache while we work on this one
        if (jt + 1 < nk)
            __builtin_prefetch(Kb + 32 * DH, 0, 3);
        v16h b00 = load_b16(Kb, DH);
        v16h b01 = load_b16(Kb + 32, DH);
        v16h b10 = load_b16(Kb + 16 * DH, DH);
        v16h b11 = load_b16(Kb + 16 * DH + 32, DH);
        v8f z = {};
        v8f s0 = wmma16(a0, b00, z); s0 = wmma16(a1, b01, s0);
        v8f s1 = wmma16(a0, b10, z); s1 = wmma16(a1, b11, s1);

        // issue V-fragment loads NOW: they fly during the softmax VALU work
        const _Float16* Vb = Vt + (size_t)bh * DH * TT + t0;
        v16h v0 = load_b16(Vb + 0 * 16 * TT, TT);
        v16h v1 = load_b16(Vb + 1 * 16 * TT, TT);
        v16h v2 = load_b16(Vb + 2 * 16 * TT, TT);
        v16h v3 = load_b16(Vb + 3 * 16 * TT, TT);

        if (t0 + 31 > q0) {           // diagonal tiles: causal mask
#pragma unroll
            for (int r = 0; r < 8; ++r) {
                int q = q0 + r + 8 * lh;
                if (t0 + n > q)       s0[r] = -1e30f;
                if (t0 + 16 + n > q)  s1[r] = -1e30f;
            }
        }
        // online softmax (row stats reduced across 16-lane halves)
#pragma unroll
        for (int r = 0; r < 8; ++r) {
            float loc = fmaxf(s0[r], s1[r]);
            loc = fmaxf(loc, __shfl_xor(loc, 1, 32));
            loc = fmaxf(loc, __shfl_xor(loc, 2, 32));
            loc = fmaxf(loc, __shfl_xor(loc, 4, 32));
            loc = fmaxf(loc, __shfl_xor(loc, 8, 32));
            float mn = fmaxf(mrow[r], loc);
            float alpha = __expf(mrow[r] - mn);
            mrow[r] = mn;
            float p0 = __expf(s0[r] - mn);
            float p1 = __expf(s1[r] - mn);
            float rs = p0 + p1;
            rs += __shfl_xor(rs, 1, 32);
            rs += __shfl_xor(rs, 2, 32);
            rs += __shfl_xor(rs, 4, 32);
            rs += __shfl_xor(rs, 8, 32);
            lrow[r] = lrow[r] * alpha + rs;
            o0[r] *= alpha; o1[r] *= alpha; o2[r] *= alpha; o3[r] *= alpha;
            int row = r + 8 * lh;
            myp[row * 32 + n]      = (_Float16)p0;   // C-layout -> LDS
            myp[row * 32 + 16 + n] = (_Float16)p1;
        }
        asm volatile("s_wait_dscnt 0" ::: "memory");
        v16h pa = load_a16(myp, 32);                 // LDS -> A-layout

        o0 = wmma16(pa, v0, o0);
        o1 = wmma16(pa, v1, o1);
        o2 = wmma16(pa, v2, o2);
        o3 = wmma16(pa, v3, o3);
    }

    int b = bh >> 4, h = bh & 15;
    v8f oo[4] = { o0, o1, o2, o3 };
#pragma unroll
    for (int r = 0; r < 8; ++r) {
        float inv = 1.0f / lrow[r];
        int q = q0 + r + 8 * lh;
        size_t base = ((size_t)(b * TT + q)) * CC + h * DH;
#pragma unroll
        for (int nt = 0; nt < 4; ++nt)
            Oh[base + nt * 16 + n] = (_Float16)(oo[nt][r] * inv);
    }
}

// ---------------- Stage 3: output projection + bias ----------------
// 32x64 tile per wave, ping-pong double buffered, fp32 out.
__global__ void __launch_bounds__(256, 1)
proj_gemm(const _Float16* __restrict__ Oh,
          const _Float16* __restrict__ wph,
          const float* __restrict__ bias,
          float* __restrict__ out) {
    int wid = blockIdx.x * 8 + (threadIdx.x >> 5);
    int mtile = wid & 255;          // 256 M tiles of 32 rows
    int n64   = wid >> 8;           // 0..15
    const _Float16* arow = Oh  + (size_t)(mtile * 32) * CC;
    const _Float16* brow = wph + (size_t)(n64 * 64) * CC;

    v8f acc[8];
#pragma unroll
    for (int i = 0; i < 8; ++i) acc[i] = (v8f){};

    FragSet s0 = load_set(arow, brow, 0);
#pragma unroll 1
    for (int k0 = 0; k0 < CC - 64; k0 += 64) {
        FragSet s1 = load_set(arow, brow, k0 + 32);
        GEMM_STEP(s0);
        s0 = load_set(arow, brow, k0 + 64);
        GEMM_STEP(s1);
    }
    {
        FragSet s1 = load_set(arow, brow, CC - 32);
        GEMM_STEP(s0);
        GEMM_STEP(s1);
    }

    int lane = threadIdx.x & 31, n = lane & 15, lh = lane >> 4;
#pragma unroll
    for (int mi = 0; mi < 2; ++mi) {
#pragma unroll
        for (int nt = 0; nt < 4; ++nt) {
            int nn = n64 * 64 + nt * 16 + n;
            float bv = bias[nn];
#pragma unroll
            for (int r = 0; r < 8; ++r) {
                int m = mtile * 32 + mi * 16 + r + 8 * lh;
                out[(size_t)m * CC + nn] = acc[mi * 4 + nt][r] + bv;
            }
        }
    }
}

extern "C" void kernel_launch(void* const* d_in, const int* in_sizes, int n_in,
                              void* d_out, int out_size, void* d_ws, size_t ws_size,
                              hipStream_t stream) {
    const float* x      = (const float*)d_in[0];
    const float* w_qkv  = (const float*)d_in[1];
    const float* w_proj = (const float*)d_in[2];
    const float* b_proj = (const float*)d_in[3];
    float* out = (float*)d_out;

    _Float16* ws = (_Float16*)d_ws;
    size_t o = 0;
    _Float16* xh  = ws + o; o += (size_t)BT * CC;   // 8.39M
    _Float16* wqh = ws + o; o += (size_t)C3 * CC;   // 3.15M
    _Float16* wph = ws + o; o += (size_t)CC * CC;   // 1.05M
    _Float16* Qh  = ws + o; o += (size_t)BT * CC;
    _Float16* Kh  = ws + o; o += (size_t)BT * CC;
    _Float16* Vt  = ws + o; o += (size_t)BT * CC;
    _Float16* Oh  = ws + o; o += (size_t)BT * CC;

    // Stage 0: converts
    {
        int n0 = BT * CC, n1 = C3 * CC, n2 = CC * CC;
        cvt_f32_f16<<<(n0 + 255) / 256, 256, 0, stream>>>(x, xh, n0);
        cvt_f32_f16<<<(n1 + 255) / 256, 256, 0, stream>>>(w_qkv, wqh, n1);
        cvt_f32_f16<<<(n2 + 255) / 256, 256, 0, stream>>>(w_proj, wph, n2);
    }
    // Stage 1: QKV GEMM — 256 M-tiles * 48 N64-tiles / 8 waves = 1536 blocks
    qkv_gemm<<<1536, 256, 0, stream>>>(xh, wqh, Qh, Kh, Vt);
    // Stage 2: attention — 64 (B*H) * 128 q-tiles / 8 waves = 1024 blocks
    attn_fwd<<<1024, 256, 0, stream>>>(Qh, Kh, Vt, Oh);
    // Stage 3: projection — 256 * 16 / 8 = 512 blocks
    proj_gemm<<<512, 256, 0, stream>>>(Oh, wph, b_proj, out);
}
